// SymHQLinear_30021821399185
// MI455X (gfx1250) — matrix operand order repaired and checked by reading, not compiled, not run on hardware
//
#include <hip/hip_runtime.h>

// ---------------------------------------------------------------------------
// SymHQLinear for MI455X (gfx1250):
//   out[8192,4096] = x[8192,4096] @ W^T,  W dequantized from 8-wide codebook
// Pass 1: dequantize W -> bf16 into d_ws (32 MB)
// Pass 2: double-buffered tiled GEMM, v_wmma_f32_16x16x32_bf16 (f32 accum),
//         B tiles staged by the Tensor Data Mover (tensor_load_to_lds +
//         s_wait_tensorcnt) when the toolchain exposes the builtin.
// ---------------------------------------------------------------------------

typedef __attribute__((ext_vector_type(16))) __bf16 v16bf;
typedef __attribute__((ext_vector_type(8)))  float  v8f;
typedef unsigned int uint32x4 __attribute__((ext_vector_type(4)));
typedef int          int32x4  __attribute__((ext_vector_type(4)));
typedef int          int32x8  __attribute__((ext_vector_type(8)));

constexpr int kTokens = 8192;
constexpr int kIn     = 4096;
constexpr int kOut    = 4096;
constexpr int kGroups = (kOut * kIn) / 8;   // 2,097,152 codebook groups

#if __has_builtin(__builtin_amdgcn_tensor_load_to_lds)
#define HAVE_TDM 1
#else
#define HAVE_TDM 0
#endif

// ---------------------------------------------------------------------------
// Kernel 1: dequantize quantized weights to bf16.
// group g covers W row o = g/512, cols [8*(g%512) .. +8).
// signs: np.packbits big-endian -> bit (7-j) is element j; bit==1 -> +1.
// ---------------------------------------------------------------------------
__global__ __launch_bounds__(256) void dequant_w_bf16(
    const float* __restrict__ codebook,   // [4096][8] f32
    const float* __restrict__ scales,     // [4096] f32
    const int*   __restrict__ indices,    // [kGroups] i32
    const int*   __restrict__ signs,      // [kGroups] i32 (one byte each)
    __bf16*      __restrict__ wq)         // [kOut][kIn] bf16 out
{
  const int g = blockIdx.x * 256 + threadIdx.x;
  const int o = g >> 9;                   // (g*8) / 4096
  const float s  = scales[o];
  const int idx  = indices[g];
  const int sb   = signs[g];
  const float4* cb = (const float4*)(codebook + idx * 8);
  const float4 c0 = cb[0], c1 = cb[1];
  const float v[8] = {c0.x, c0.y, c0.z, c0.w, c1.x, c1.y, c1.z, c1.w};
  union { __bf16 h[8]; uint4 q; } r;
#pragma unroll
  for (int j = 0; j < 8; ++j) {
    const float f = v[j] * s;
    r.h[j] = (__bf16)(((sb >> (7 - j)) & 1) ? f : -f);
  }
  *(uint4*)(wq + (size_t)g * 8) = r.q;
}

// ---------------------------------------------------------------------------
// GEMM configuration
// Block: 256 threads (8 waves), 128x128 output tile, BK=64, double-buffered.
// Wave grid 2(M) x 4(N): each wave -> 64x32 = 4x2 tiles of 16x16.
// ---------------------------------------------------------------------------
constexpr int BM  = 128;
constexpr int BN  = 128;
constexpr int BK  = 64;
constexpr int LDT = BK + 8;   // LDS row stride (bf16): 144 B = 16B-aligned, 36-bank stride

#if HAVE_TDM
// TDM: async-copy a 2D bf16 tile [tile_rows x BK] from a row-major
// [kOut x kIn] bf16 tensor into LDS, padding each 128 B row with 16 B so the
// LDS row stride is 144 B (= LDT bf16), matching the WMMA fragment reads.
// Descriptor layout per CDNA5 ISA.
__device__ __forceinline__ void tdm_load_b_tile(unsigned int lds_byte_off,
                                                const void* gaddr,
                                                int tile_rows) {
  const unsigned long long ga = (unsigned long long)(uintptr_t)gaddr;
  uint32x4 g0;
  g0.x = 1u;                                   // count=1, user descriptor
  g0.y = lds_byte_off;                         // lds_addr
  g0.z = (unsigned int)ga;                     // global_addr[31:0]
  g0.w = (unsigned int)(ga >> 32) | 0x80000000u; // global_addr[56:32] | type=2<<30
  int32x8 g1;
  g1[0] = (1 << 16)        // data_size = 2 bytes
        | (1 << 20)        // pad_enable (LDS destination padding)
        | (4 << 22)        // pad_interval: every 32 dwords (= 128 B row)
        | (3 << 25);       // pad_amount: 4 dwords (= 16 B) -> 144 B LDS stride
  g1[1] = (int)((kIn & 0xFFFF) << 16);                          // tensor_dim0 lo
  g1[2] = (int)(((kIn >> 16) & 0xFFFF) | ((kOut & 0xFFFF) << 16)); // dim0 hi | dim1 lo
  g1[3] = (int)(((kOut >> 16) & 0xFFFF) | (BK << 16));          // dim1 hi | tile_dim0
  g1[4] = tile_rows;                                            // tile_dim1 (tile_dim2=0)
  g1[5] = kIn;                                                  // tensor_dim0_stride lo
  g1[6] = 0;                                                    // stride hi | dim1_stride lo
  g1[7] = 0;
  const int32x4 z4 = {0, 0, 0, 0};
#if defined(__clang_major__) && (__clang_major__ >= 23)
  const int32x8 z8 = {0, 0, 0, 0, 0, 0, 0, 0};
  __builtin_amdgcn_tensor_load_to_lds(g0, g1, z4, z4, z8, 0);
#else
  __builtin_amdgcn_tensor_load_to_lds(g0, g1, z4, z4, 0);
#endif
}

__device__ __forceinline__ void wait_tensorcnt0() {
#if __has_builtin(__builtin_amdgcn_s_wait_tensorcnt)
  __builtin_amdgcn_s_wait_tensorcnt(0);
#else
  asm volatile("s_wait_tensorcnt 0x0" ::: "memory");
#endif
}
#endif  // HAVE_TDM

// ---------------------------------------------------------------------------
// Kernel 2: GEMM out = x @ wq^T via WMMA bf16, double-buffered LDS.
// ---------------------------------------------------------------------------
__global__ __launch_bounds__(256) void qgemm_wmma_bf16(
    const float*  __restrict__ x,    // [kTokens][kIn] f32
    const __bf16* __restrict__ wq,   // [kOut][kIn] bf16
    float*        __restrict__ out)  // [kTokens][kOut] f32
{
  __shared__ __align__(16) __bf16 lA[2][BM][LDT];   // x tiles   (rows = tokens)
  __shared__ __align__(16) __bf16 lB[2][BN][LDT];   // W tiles   (rows = out ch)

  const int tid  = threadIdx.x;
  const int lane = tid & 31;
  const int wave = tid >> 5;
  const int bm   = blockIdx.y * BM;
  const int bn   = blockIdx.x * BN;
  const int waveM = (wave >> 2) * 64;   // 0 or 64
  const int waveN = (wave & 3) * 32;    // 0,32,64,96

  const int halfSel = (lane < 16) ? 0 : 1;   // lane-half select per ISA layouts
  const int lm      = lane & 15;

  v8f acc[4][2] = {};   // f32 accumulators, 4 M-tiles x 2 N-tiles

  // staging coordinates
  const int arow = tid >> 4;            // A: 16 rows/pass, 16 float4 per row
  const int acol = (tid & 15) * 4;      //    f32 column
  const int brow = tid >> 3;            // B: 32 rows/pass (fallback path)
  const int bcol = (tid & 7) * 8;       //    bf16 column (8 bf16 = 16 B)

  const int NK = kIn / BK;              // 64 K-steps

  // ---- prologue: stage tile 0 into buffer 0 ----
  {
#pragma unroll
    for (int r = 0; r < 8; ++r) {
      const int row = arow + r * 16;
      const float4 f = *(const float4*)(x + (size_t)(bm + row) * kIn + acol);
      union { __bf16 h[4]; uint2 u; } pk;
      pk.h[0] = (__bf16)f.x; pk.h[1] = (__bf16)f.y;
      pk.h[2] = (__bf16)f.z; pk.h[3] = (__bf16)f.w;
      *(uint2*)&lA[0][row][acol] = pk.u;
    }
#if HAVE_TDM
    if (wave == 0) {
      tdm_load_b_tile((unsigned int)(size_t)&lB[0][0][0],
                      wq + (size_t)bn * kIn, BN);
      wait_tensorcnt0();
    }
#else
#pragma unroll
    for (int r = 0; r < 4; ++r) {
      const int row = brow + r * 32;
      *(uint4*)&lB[0][row][bcol] =
          *(const uint4*)(wq + (size_t)(bn + row) * kIn + bcol);
    }
#endif
  }
  __syncthreads();

  // ---- main loop: compute buffer p while staging buffer 1-p ----
  for (int kt = 0; kt < NK; ++kt) {
    const int p   = kt & 1;
    const int kn  = (kt + 1) * BK;       // K offset of the *next* tile
    const bool pre = (kt + 1 < NK);

    // prefetch next A tile into registers; kick TDM for next B tile
    float4 af[8];
#if !HAVE_TDM
    uint4 bq[4];
#endif
    if (pre) {
#pragma unroll
      for (int r = 0; r < 8; ++r)
        af[r] = *(const float4*)(x + (size_t)(bm + arow + r * 16) * kIn + kn + acol);
#if HAVE_TDM
      if (wave == 0)
        tdm_load_b_tile((unsigned int)(size_t)&lB[1 - p][0][0],
                        wq + (size_t)bn * kIn + kn, BN);
#else
#pragma unroll
      for (int r = 0; r < 4; ++r)
        bq[r] = *(const uint4*)(wq + (size_t)(bn + brow + r * 32) * kIn + kn + bcol);
#endif
    }

    // ---- two K=32 WMMA steps on buffer p ----
#pragma unroll
    for (int kc = 0; kc < 2; ++kc) {
      const int kb = kc * 32;
      union FragA { v16bf v; uint4 q[2]; } a[4];
      union FragB { v16bf v; uint4 q[2]; } b[2];
      // A frag: lane half 0 -> K {0..7,16..23}; half 1 -> K {8..15,24..31}
#pragma unroll
      for (int mt = 0; mt < 4; ++mt) {
        const __bf16* src = &lA[p][waveM + mt * 16 + lm][kb + halfSel * 8];
        a[mt].q[0] = *(const uint4*)(src);
        a[mt].q[1] = *(const uint4*)(src + 16);
      }
      // B frag: col N = lane%16; lane half 0 -> K {0..7,16..23}; half 1 -> K {8..15,24..31}
#pragma unroll
      for (int nt = 0; nt < 2; ++nt) {
        const __bf16* src = &lB[p][waveN + nt * 16 + lm][kb + halfSel * 8];
        b[nt].q[0] = *(const uint4*)(src);
        b[nt].q[1] = *(const uint4*)(src + 16);
      }
#pragma unroll
      for (int mt = 0; mt < 4; ++mt)
#pragma unroll
        for (int nt = 0; nt < 2; ++nt)
          acc[mt][nt] = __builtin_amdgcn_wmma_f32_16x16x32_bf16(
              false, a[mt].v, false, b[nt].v, (short)0, acc[mt][nt],
              false, false);
    }

    // write staged registers to the other buffer (its readers finished at the
    // barrier ending iteration kt-1)
    if (pre) {
#pragma unroll
      for (int r = 0; r < 8; ++r) {
        union { __bf16 h[4]; uint2 u; } pk;
        pk.h[0] = (__bf16)af[r].x; pk.h[1] = (__bf16)af[r].y;
        pk.h[2] = (__bf16)af[r].z; pk.h[3] = (__bf16)af[r].w;
        *(uint2*)&lA[1 - p][arow + r * 16][acol] = pk.u;
      }
#if HAVE_TDM
      if (wave == 0) wait_tensorcnt0();   // next B tile landed in LDS
#else
#pragma unroll
      for (int r = 0; r < 4; ++r)
        *(uint4*)&lB[1 - p][brow + r * 32][bcol] = bq[r];
#endif
    }
    __syncthreads();
  }

  // ---- epilogue: D 16x16 f32 layout: VGPR r -> M = r (+8 upper lanes), N = lane%16
#pragma unroll
  for (int mt = 0; mt < 4; ++mt) {
#pragma unroll
    for (int nt = 0; nt < 2; ++nt) {
      const int col = bn + waveN + nt * 16 + lm;
#pragma unroll
      for (int r = 0; r < 8; ++r) {
        const int row = bm + waveM + mt * 16 + r + halfSel * 8;
        out[(size_t)row * kOut + col] = acc[mt][nt][r];
      }
    }
  }
}

// ---------------------------------------------------------------------------
extern "C" void kernel_launch(void* const* d_in, const int* in_sizes, int n_in,
                              void* d_out, int out_size, void* d_ws, size_t ws_size,
                              hipStream_t stream) {
  const float* x        = (const float*)d_in[0];   // [8192,4096] f32
  const float* codebook = (const float*)d_in[1];   // [4096,8]    f32
  const float* scales   = (const float*)d_in[2];   // [4096,1]    f32
  const int*   indices  = (const int*)d_in[3];     // [2M]        i32
  const int*   signs    = (const int*)d_in[4];     // [2M]        i32 (bytes)
  float*       out      = (float*)d_out;           // [8192,4096] f32
  __bf16*      wq       = (__bf16*)d_ws;           // 32 MB scratch, fully rewritten each call

  dequant_w_bf16<<<kGroups / 256, 256, 0, stream>>>(codebook, scales, indices, signs, wq);

  dim3 grid(kOut / BN, kTokens / BM);  // 32 x 64 blocks
  qgemm_wmma_bf16<<<grid, 256, 0, stream>>>(x, wq, out);
}
